// RotationComponent_979252544004
// MI455X (gfx1250) — matrix operand order repaired and checked by reading, not compiled, not run on hardware
//
#include <hip/hip_runtime.h>

#define DIM   4096
#define KT    32            // K-tile
#define LDAH  40            // LDS row pitch in halves: 32 data + 8 pad (80B rows, 16B aligned)
#define BM    128           // block tile M
#define BN    128           // block tile N
#define NKS   (DIM / KT)    // 128 K-steps

typedef __attribute__((ext_vector_type(16))) __bf16          v16bf;
typedef __attribute__((ext_vector_type(8)))  __bf16          v8bf;
typedef __attribute__((ext_vector_type(8)))  float           v8f;
typedef __attribute__((ext_vector_type(4)))  float           v4f;
typedef __attribute__((ext_vector_type(2)))  unsigned        ui2;

// Round-to-nearest-even fp32 -> bf16 (used only for the lo residual).
static __device__ __forceinline__ unsigned short f2bf_rn(float f) {
  unsigned u = __float_as_uint(f);
  u += 0x7FFFu + ((u >> 16) & 1u);
  return (unsigned short)(u >> 16);
}
// Truncating hi-split of a float pair packed into one dword: one v_perm_b32.
// Result[15:0] = bf16_trunc(lo_elem), Result[31:16] = bf16_trunc(hi_elem).
static __device__ __forceinline__ unsigned pack_bf_trunc(float lo_elem, float hi_elem) {
  return __builtin_amdgcn_perm(__float_as_uint(hi_elem), __float_as_uint(lo_elem),
                               0x07060302u);
}
// float value of the truncated hi part: one v_and_b32.
static __device__ __forceinline__ float hi_part(float f) {
  return __uint_as_float(__float_as_uint(f) & 0xFFFF0000u);
}
static __device__ __forceinline__ unsigned pack2(unsigned short a, unsigned short b) {
  return (unsigned)a | ((unsigned)b << 16);
}
// Concatenate two 8x bf16 LDS chunks (ds_load_b128 each) into one 16-elem fragment.
static __device__ __forceinline__ v16bf frag16(const unsigned short* p0,
                                               const unsigned short* p1) {
  v8bf a = *(const v8bf*)p0;
  v8bf b = *(const v8bf*)p1;
  return __builtin_shufflevector(a, b, 0,1,2,3,4,5,6,7,8,9,10,11,12,13,14,15);
}

__global__ __launch_bounds__(256)
void rot_gemm_bf16x3(const float* __restrict__ W, const float* __restrict__ X,
                     const float* __restrict__ R, float* __restrict__ out)
{
  // Double-buffered hi/lo bf16 tiles: A row-major [M][K], B transposed [N][K].
  __shared__ __align__(16) unsigned short sAh[2][BM * LDAH];
  __shared__ __align__(16) unsigned short sAl[2][BM * LDAH];
  __shared__ __align__(16) unsigned short sBh[2][BN * LDAH];
  __shared__ __align__(16) unsigned short sBl[2][BN * LDAH];

  const int tid  = threadIdx.x;
  const int lane = tid & 31;
  const int wave = tid >> 5;      // 0..7
  const int wm   = wave >> 1;     // 0..3 -> 32 rows each
  const int wn   = wave & 1;      // 0..1 -> 64 cols each
  const int m0w  = wm * 32;
  const int n0w  = wn * 64;
  const int l16  = lane & 15;
  const int lhi  = lane >> 4;     // 0/1 half-wave

  const int bn = blockIdx.x;      // 0..31  (N blocks fast -> R stays in L2)
  const int bm = blockIdx.y;      // 0..95  (rows 0..31 -> W, 32..95 -> X)

  const float* Asrc;
  float*       Odst;
  if (bm < 32) {
    Asrc = W + (size_t)bm * BM * DIM;
    Odst = out + (size_t)bm * BM * DIM;
  } else {
    Asrc = X + (size_t)(bm - 32) * BM * DIM;
    Odst = out + (size_t)DIM * DIM + (size_t)(bm - 32) * BM * DIM;
  }
  const float* Bsrc = R + (size_t)bn * BN;

  v4f   aReg[4];
  float b0Reg[8], b1Reg[8];
  v8f   acc[2][4] = {};

  // ---- prologue: load k-step 0 into registers ----
#pragma unroll
  for (int i = 0; i < 4; ++i) {
    int q = i * 256 + tid; int row = q >> 3; int c4 = (q & 7) << 2;
    aReg[i] = *(const v4f*)(Asrc + (size_t)row * DIM + c4);
  }
#pragma unroll
  for (int i = 0; i < 8; ++i) {
    int q = i * 256 + tid; int kp = q >> 7; int n = q & 127;
    const float* p = Bsrc + (size_t)(2 * kp) * DIM + n;
    b0Reg[i] = p[0];
    b1Reg[i] = p[DIM];
  }

  int buf = 0;
  for (int ks = 0; ks < NKS; ++ks) {
    // ---- stash: fp32 regs -> hi(trunc)/lo(RN residual) bf16 in LDS ----
#pragma unroll
    for (int i = 0; i < 4; ++i) {
      int q = i * 256 + tid; int row = q >> 3; int c4 = (q & 7) << 2;
      float f0 = aReg[i][0], f1 = aReg[i][1], f2 = aReg[i][2], f3 = aReg[i][3];
      ui2 h, l;
      h[0] = pack_bf_trunc(f0, f1);
      h[1] = pack_bf_trunc(f2, f3);
      l[0] = pack2(f2bf_rn(f0 - hi_part(f0)), f2bf_rn(f1 - hi_part(f1)));
      l[1] = pack2(f2bf_rn(f2 - hi_part(f2)), f2bf_rn(f3 - hi_part(f3)));
      *(ui2*)(&sAh[buf][row * LDAH + c4]) = h;
      *(ui2*)(&sAl[buf][row * LDAH + c4]) = l;
    }
#pragma unroll
    for (int i = 0; i < 8; ++i) {
      int q = i * 256 + tid; int kp = q >> 7; int n = q & 127;
      float f0 = b0Reg[i], f1 = b1Reg[i];   // K even in low half
      *(unsigned*)(&sBh[buf][n * LDAH + 2 * kp]) = pack_bf_trunc(f0, f1);
      *(unsigned*)(&sBl[buf][n * LDAH + 2 * kp]) =
          pack2(f2bf_rn(f0 - hi_part(f0)), f2bf_rn(f1 - hi_part(f1)));
    }

    __syncthreads();

    // ---- prefetch next k-step while this one computes ----
    if (ks + 1 < NKS) {
      int k0 = (ks + 1) * KT;
#pragma unroll
      for (int i = 0; i < 4; ++i) {
        int q = i * 256 + tid; int row = q >> 3; int c4 = (q & 7) << 2;
        aReg[i] = *(const v4f*)(Asrc + (size_t)row * DIM + k0 + c4);
      }
#pragma unroll
      for (int i = 0; i < 8; ++i) {
        int q = i * 256 + tid; int kp = q >> 7; int n = q & 127;
        const float* p = Bsrc + (size_t)(k0 + 2 * kp) * DIM + n;
        b0Reg[i] = p[0];
        b1Reg[i] = p[DIM];
      }
    }

    // ---- compute: 2 M-tiles x 4 N-tiles x 3 WMMAs (bf16x3) ----
    v16bf afh[2], afl[2];
#pragma unroll
    for (int t = 0; t < 2; ++t) {
      const unsigned short* ph = &sAh[buf][(m0w + t * 16 + l16) * LDAH + lhi * 16];
      const unsigned short* pl = &sAl[buf][(m0w + t * 16 + l16) * LDAH + lhi * 16];
      afh[t] = frag16(ph, ph + 8);    // K chunks 0..7 / 8..15 (half-wave +16)
      afl[t] = frag16(pl, pl + 8);
    }
    // B fragments double-buffered in registers to decouple ds latency from WMMA.
    const int brow = (n0w + l16) * LDAH + lhi * 16;
    v16bf bh = frag16(&sBh[buf][brow], &sBh[buf][brow] + 8);
    v16bf bl = frag16(&sBl[buf][brow], &sBl[buf][brow] + 8);
#pragma unroll
    for (int j = 0; j < 4; ++j) {
      v16bf bhN = {}, blN = {};
      if (j < 3) {
        int r = brow + (j + 1) * 16 * LDAH;
        bhN = frag16(&sBh[buf][r], &sBh[buf][r] + 8);
        blN = frag16(&sBl[buf][r], &sBl[buf][r] + 8);
      }
#pragma unroll
      for (int t = 0; t < 2; ++t) {
        acc[t][j] = __builtin_amdgcn_wmma_f32_16x16x32_bf16(
            false, afh[t], false, bh, (short)0, acc[t][j], false, false);
        acc[t][j] = __builtin_amdgcn_wmma_f32_16x16x32_bf16(
            false, afl[t], false, bh, (short)0, acc[t][j], false, false);
        acc[t][j] = __builtin_amdgcn_wmma_f32_16x16x32_bf16(
            false, afh[t], false, bl, (short)0, acc[t][j], false, false);
      }
      bh = bhN; bl = blN;
    }
    buf ^= 1;
  }

  // ---- epilogue: C/D layout -> global (non-temporal, keep R in L2) ----
#pragma unroll
  for (int t = 0; t < 2; ++t) {
    int row0 = m0w + t * 16 + lhi * 8;
#pragma unroll
    for (int j = 0; j < 4; ++j) {
      float* o = Odst + (size_t)row0 * DIM + (size_t)(bn * BN + n0w + j * 16 + l16);
#pragma unroll
      for (int r = 0; r < 8; ++r)
        __builtin_nontemporal_store(acc[t][j][r], o + (size_t)r * DIM);
    }
  }
}

extern "C" void kernel_launch(void* const* d_in, const int* in_sizes, int n_in,
                              void* d_out, int out_size, void* d_ws, size_t ws_size,
                              hipStream_t stream) {
  const float* W = (const float*)d_in[0];   // [4096, 4096]
  const float* X = (const float*)d_in[1];   // [4, 2048, 4096] == [8192, 4096]
  const float* R = (const float*)d_in[2];   // [4096, 4096]
  float* out = (float*)d_out;               // w_rot (16M floats) then a_rot (32M floats)
  (void)in_sizes; (void)n_in; (void)out_size; (void)d_ws; (void)ws_size;

  dim3 grid(DIM / BN, (3 * DIM) / BM);      // (32, 96): M_total = 12288
  rot_gemm_bf16x3<<<grid, 256, 0, stream>>>(W, X, R, out);
}
